// MultiHeadAttention_33818572488791
// MI455X (gfx1250) — matrix operand order repaired and checked by reading, not compiled, not run on hardware
//
#include <hip/hip_runtime.h>
#include <hip/hip_bf16.h>
#include <stdint.h>

// MI455X (gfx1250) flash-attention MHA, bf16 WMMA with f32 accumulation.
// S=2048, B=2, D=1024, H=16, E=64. ~70 GFLOP total; flash formulation avoids
// the 537MB score tensor so the whole pipeline is ~100-150MB of HBM traffic
// (~6us at 23.3 TB/s). K/V tiles are double-buffered in LDS and fetched with
// gfx1250 async global->LDS copies (ASYNCcnt) so tile fetch overlaps WMMA.

#define S_LEN  2048
#define BATCH  2
#define DMODEL 1024
#define NHEAD  16
#define EHEAD  64
#define MASKV  1e18f

typedef __attribute__((ext_vector_type(16))) __bf16 v16bf;
typedef __attribute__((ext_vector_type(8)))  __bf16 v8bf;
typedef __attribute__((ext_vector_type(8)))  float  v8f;

// The async-LDS builtin takes generic-vector int4 pointers qualified with
// address spaces (HIP prints AS1 as __device__, AS3 as __shared__).
typedef int v4i_vs __attribute__((vector_size(16)));
typedef __attribute__((address_space(1))) void*    as1_void_t;
typedef __attribute__((address_space(3))) void*    as3_void_t;
typedef __attribute__((address_space(1))) v4i_vs*  as1_v4i_t;
typedef __attribute__((address_space(3))) v4i_vs*  as3_v4i_t;

#if defined(__HIP_DEVICE_COMPILE__) && __has_builtin(__builtin_amdgcn_global_load_async_to_lds_b128)
#define ASYNC_LDS 1
#else
#define ASYNC_LDS 0
#endif

// 16-byte global -> LDS copy; async path uses the gfx1250 async-DMA engine
// (tracked by ASYNCcnt), fallback is a synchronous vector copy.
__device__ __forceinline__ void copy16_g2l(const __bf16* g, __bf16* l) {
#if ASYNC_LDS
  __builtin_amdgcn_global_load_async_to_lds_b128(
      (as1_v4i_t)(as1_void_t)g,
      (as3_v4i_t)(as3_void_t)l, 0, 0);
#else
  *(v8bf*)l = *(const v8bf*)g;
#endif
}

__device__ __forceinline__ void wait_async_le2() {
  asm volatile("s_wait_asynccnt 0x2" ::: "memory");
}
__device__ __forceinline__ void wait_async_0() {
  asm volatile("s_wait_asynccnt 0x0" ::: "memory");
}

__device__ __forceinline__ v16bf cat8(v8bf lo, v8bf hi) {
  return __builtin_shufflevector(lo, hi, 0,1,2,3,4,5,6,7,8,9,10,11,12,13,14,15);
}

// A-matrix fragment (16x32 bf16, MxK): lane r (0-15) = row M=r; per ISA table
// the per-lane K runs are [half*8 .. half*8+7] and [16+half*8 .. 16+half*8+7].
__device__ __forceinline__ v16bf load_fragA(const __bf16* tile, int ld) {
  const int lane = threadIdx.x & 31;
  const __bf16* p = tile + (lane & 15) * ld + (lane >> 4) * 8;
  return cat8(*(const v8bf*)p, *(const v8bf*)(p + 16));
}

// B-matrix fragment (32x16 bf16, KxN), sourced from N-major (B^T row-major)
// storage: lane r = column N=r, K runs [half*8 .. half*8+7] and [16+half*8 .. 16+half*8+7].
__device__ __forceinline__ v16bf load_fragB(const __bf16* tileT, int ld) {
  const int lane = threadIdx.x & 31;
  const __bf16* p = tileT + (lane & 15) * ld + (lane >> 4) * 8;
  return cat8(*(const v8bf*)p, *(const v8bf*)(p + 16));
}

__device__ __forceinline__ v8f wmma_bf16(v16bf a, v16bf b, v8f c) {
  return __builtin_amdgcn_wmma_f32_16x16x32_bf16(false, a, false, b,
                                                 (short)0, c, false, false);
}

__device__ __forceinline__ v8f zero8() {
  v8f z = {0.f,0.f,0.f,0.f,0.f,0.f,0.f,0.f};
  return z;
}

// key_mask[j,m] (numpy bool, 1 byte) -> additive bias mb[m*S + j] in {-1e18, 0}
__global__ void maskbias_kernel(const unsigned char* __restrict__ km,
                                float* __restrict__ mb) {
  const int idx = blockIdx.x * blockDim.x + threadIdx.x;
  if (idx < S_LEN * BATCH) {
    const int m = idx / S_LEN;
    const int j = idx - m * S_LEN;
    mb[idx] = km[(size_t)j * BATCH + m] ? -MASKV : 0.0f;
  }
}

// D[row][col] = (sum_k A[row][k] * W[col][k] + bias[col]) * scale
// A: [4096 x 1024] f32 or bf16; W: [1024 x 1024] f32 (row = output channel).
// MODE 0: scatter bf16 into QK layout [h][m][S][64]   (row=i*B+m, col=h*64+e)
// MODE 2: scatter bf16 into V^T layout [h][m][64][S]  (transposed for async)
// MODE 1: dense f32 row-major [row][col]
template <bool A_IS_BF16, int MODE>
__global__ __launch_bounds__(256) void gemm_bf16_wmma(
    const void* __restrict__ Aptr, const float* __restrict__ W,
    const float* __restrict__ bias, void* __restrict__ Dst, float scale)
{
  __shared__ __bf16 Asm[128 * 40];  // 128 rows x 32 k (ld 40, 16B aligned rows)
  __shared__ __bf16 Bsm[64 * 40];   // 64 cols x 32 k

  const int tid  = threadIdx.x;
  const int lane = tid & 31;
  const int wave = tid >> 5;
  const int wm = wave & 3;          // 4 M-slots of 32 rows
  const int wn = wave >> 2;         // 2 N-slots of 32 cols
  const int r = lane & 15;
  const int half = lane >> 4;
  const int mBase = blockIdx.y * 128;
  const int nBase = blockIdx.x * 64;

  v8f c00 = zero8(), c01 = zero8(), c10 = zero8(), c11 = zero8();

  const int arow = tid >> 1, ak0 = (tid & 1) * 16;  // 16 elems/thread for A
  const int brow = tid >> 2, bk0 = (tid & 3) * 8;   // 8 elems/thread for B

  for (int kb = 0; kb < DMODEL; kb += 32) {
    __syncthreads();
    if constexpr (A_IS_BF16) {
      const __bf16* src = (const __bf16*)Aptr + (size_t)(mBase + arow) * DMODEL + kb + ak0;
      *(v8bf*)&Asm[arow * 40 + ak0]     = *(const v8bf*)src;
      *(v8bf*)&Asm[arow * 40 + ak0 + 8] = *(const v8bf*)(src + 8);
    } else {
      const float* src = (const float*)Aptr + (size_t)(mBase + arow) * DMODEL + kb + ak0;
      #pragma unroll
      for (int q = 0; q < 16; ++q) Asm[arow * 40 + ak0 + q] = (__bf16)src[q];
    }
    {
      const float* wsrc = W + (size_t)(nBase + brow) * DMODEL + kb + bk0;
      #pragma unroll
      for (int q = 0; q < 8; ++q) Bsm[brow * 40 + bk0 + q] = (__bf16)wsrc[q];
    }
    if (kb + 32 < DMODEL) {
      __builtin_prefetch((const char*)Aptr + ((size_t)(mBase + arow) * DMODEL + kb + 32), 0, 0);
    }
    __syncthreads();

    v16bf a0 = load_fragA(&Asm[(wm * 32 + 0)  * 40], 40);
    v16bf a1 = load_fragA(&Asm[(wm * 32 + 16) * 40], 40);
    v16bf b0 = load_fragB(&Bsm[(wn * 32 + 0)  * 40], 40);
    v16bf b1 = load_fragB(&Bsm[(wn * 32 + 16) * 40], 40);
    c00 = wmma_bf16(a0, b0, c00);
    c01 = wmma_bf16(a0, b1, c01);
    c10 = wmma_bf16(a1, b0, c10);
    c11 = wmma_bf16(a1, b1, c11);
  }

  // C/D layout: lane half -> rows v (lo) / v+8 (hi); N = lane&15.
  const int col0 = nBase + wn * 32 + r;
  const int col1 = col0 + 16;
  const float bb0 = bias[col0], bb1 = bias[col1];
  #pragma unroll
  for (int v = 0; v < 8; ++v) {
    const int row0 = mBase + wm * 32 + v + 8 * half;
    const int row1 = row0 + 16;
    const float x00 = (c00[v] + bb0) * scale;
    const float x01 = (c01[v] + bb1) * scale;
    const float x10 = (c10[v] + bb0) * scale;
    const float x11 = (c11[v] + bb1) * scale;
    if constexpr (MODE == 0 || MODE == 2) {
      __bf16* o = (__bf16*)Dst;
      #define QKV_ST(row, col, val) do {                                         \
        const int i_ = (row) >> 1, m_ = (row) & 1;                               \
        const int h_ = (col) >> 6, e_ = (col) & 63;                              \
        const size_t idx_ = (MODE == 0)                                          \
          ? ((((size_t)h_ * BATCH + m_) * S_LEN + i_) * EHEAD + e_)              \
          : ((((size_t)h_ * BATCH + m_) * EHEAD + e_) * S_LEN + i_);             \
        o[idx_] = (__bf16)(val);                                                 \
      } while (0)
      QKV_ST(row0, col0, x00); QKV_ST(row0, col1, x01);
      QKV_ST(row1, col0, x10); QKV_ST(row1, col1, x11);
      #undef QKV_ST
    } else {
      float* o = (float*)Dst;
      o[(size_t)row0 * DMODEL + col0] = x00;
      o[(size_t)row0 * DMODEL + col1] = x01;
      o[(size_t)row1 * DMODEL + col0] = x10;
      o[(size_t)row1 * DMODEL + col1] = x11;
    }
  }
}

// Flash attention: one (h,m) per blockIdx.y; 8 waves x 16 query rows = 128 rows
// per workgroup. K/V tiles double-buffered in LDS with async global->LDS
// copies. Iterates ALL 64 j-tiles with the reference's additive masks
// (causal -1e18 and key-mask -1e18), so fully-masked rows match the reference.
__global__ __launch_bounds__(256) void attn_kernel(
    const __bf16* __restrict__ Qh, const __bf16* __restrict__ Kh,
    const __bf16* __restrict__ Vh, const float* __restrict__ mbias,
    __bf16* __restrict__ Obuf)
{
  __shared__ __bf16 Kt[2][32 * 72];     // K tile [j][e], N-major for B frags
  __shared__ __bf16 Vt[2][64 * 40];     // V^T tile [e][j], N-major for B frags
  __shared__ __bf16 Pt[8 * 16 * 40];    // per-wave P scratch [16][32] (+pad)

  const int tid  = threadIdx.x;
  const int lane = tid & 31;
  const int wave = tid >> 5;
  const int r    = lane & 15;
  const int half = lane >> 4;
  const int hm   = blockIdx.y;          // h * BATCH + m
  const int m    = hm % BATCH;
  const int h    = hm / BATCH;
  const int rowBase = blockIdx.x * 128 + wave * 16;

  const __bf16* Qb = Qh + (size_t)hm * S_LEN * EHEAD;   // [i][e]
  const __bf16* Kb = Kh + (size_t)hm * S_LEN * EHEAD;   // [j][e]
  const __bf16* Vb = Vh + (size_t)hm * EHEAD * S_LEN;   // [e][j] (pre-transposed)
  const float* mbm = mbias + (size_t)m * S_LEN;

  // cooperative tile-copy indices (256 threads, 16B each per array)
  const int kj  = tid >> 3;            // 0..31 (j row of K tile)
  const int ke0 = (tid & 7) * 8;       // 0..56 (e chunk)
  const int ve  = tid >> 2;            // 0..63 (e row of V^T tile)
  const int vj0 = (tid & 3) * 8;       // 0..24 (j chunk)

  auto issue_tile = [&](int t, int buf) {
    const int jBase = t * 32;
    copy16_g2l(Kb + (size_t)(jBase + kj) * EHEAD + ke0, &Kt[buf][kj * 72 + ke0]);
    copy16_g2l(Vb + (size_t)ve * S_LEN + jBase + vj0,   &Vt[buf][ve * 40 + vj0]);
  };

  // Q fragments (rows rowBase..rowBase+15, E=64 split into two K=32 chunks).
  const __bf16* qrow = Qb + (size_t)(rowBase + r) * EHEAD;
  const v16bf qf0 = cat8(*(const v8bf*)(qrow + half * 8),
                         *(const v8bf*)(qrow + half * 8 + 16));
  const v16bf qf1 = cat8(*(const v8bf*)(qrow + 32 + half * 8),
                         *(const v8bf*)(qrow + 32 + half * 8 + 16));

  v8f o0 = zero8(), o1 = zero8(), o2 = zero8(), o3 = zero8();
  float mloc[8], lloc[8];
  #pragma unroll
  for (int v = 0; v < 8; ++v) { mloc[v] = -1e30f; lloc[v] = 0.f; }

  const int nT = S_LEN / 32;           // 64 j-tiles
  issue_tile(0, 0);                    // prologue: async-fill buffer 0

  for (int t = 0; t < nT; ++t) {
    const int cur = t & 1;
    const int jBase = t * 32;

    __syncthreads();                   // all waves done reading buf cur^1
    if (t + 1 < nT) {
      issue_tile(t + 1, cur ^ 1);      // overlap next tile fetch with compute
      wait_async_le2();                // in-order => tile t's 2 copies landed
    } else {
      wait_async_0();
    }
    __syncthreads();                   // tile t resident for every wave

    // scores S[16 x 32] = Q[16 x 64] * K^T (two N-subtiles, two K-chunks each)
    v8f s0 = zero8(), s1 = zero8();
    {
      v16bf kA0 = load_fragB(&Kt[cur][0 * 72 + 0],  72);
      v16bf kA1 = load_fragB(&Kt[cur][0 * 72 + 32], 72);
      s0 = wmma_bf16(qf0, kA0, s0);
      s0 = wmma_bf16(qf1, kA1, s0);
      v16bf kB0 = load_fragB(&Kt[cur][16 * 72 + 0],  72);
      v16bf kB1 = load_fragB(&Kt[cur][16 * 72 + 32], 72);
      s1 = wmma_bf16(qf0, kB0, s1);
      s1 = wmma_bf16(qf1, kB1, s1);
    }

    const float mb0 = mbm[jBase + r];
    const float mb1 = mbm[jBase + 16 + r];
    const int j0 = jBase + r, j1 = jBase + 16 + r;
    __bf16* pw = &Pt[wave * 16 * 40];

    #pragma unroll
    for (int v = 0; v < 8; ++v) {
      const int i = rowBase + v + 8 * half;
      const float sv0 = s0[v] + mb0 + ((j0 > i) ? -MASKV : 0.f);
      const float sv1 = s1[v] + mb1 + ((j1 > i) ? -MASKV : 0.f);
      // row reductions across the 16 lanes holding this row's columns
      float rmx = fmaxf(sv0, sv1);
      #pragma unroll
      for (int off = 8; off >= 1; off >>= 1) rmx = fmaxf(rmx, __shfl_xor(rmx, off, 32));
      const float mnew = fmaxf(mloc[v], rmx);
      const float p0 = __expf(sv0 - mnew);
      const float p1 = __expf(sv1 - mnew);
      float rs = p0 + p1;
      #pragma unroll
      for (int off = 8; off >= 1; off >>= 1) rs += __shfl_xor(rs, off, 32);
      const float corr = __expf(mloc[v] - mnew);
      lloc[v] = lloc[v] * corr + rs;
      mloc[v] = mnew;
      o0[v] *= corr; o1[v] *= corr; o2[v] *= corr; o3[v] *= corr;
      const int prow = v + 8 * half;   // C-layout row -> P scratch row
      pw[prow * 40 + r]      = (__bf16)p0;
      pw[prow * 40 + 16 + r] = (__bf16)p1;
    }

    // wave-synchronous LDS shuffle: stores above feed A-fragment loads below
    asm volatile("s_wait_dscnt 0x0" ::: "memory");

    v16bf pf = load_fragA(pw, 40);
    o0 = wmma_bf16(pf, load_fragB(&Vt[cur][0  * 40], 40), o0);
    o1 = wmma_bf16(pf, load_fragB(&Vt[cur][16 * 40], 40), o1);
    o2 = wmma_bf16(pf, load_fragB(&Vt[cur][32 * 40], 40), o2);
    o3 = wmma_bf16(pf, load_fragB(&Vt[cur][48 * 40], 40), o3);
  }

  // epilogue: divide by l, store bf16 into [i][m][h*64+e] (concat-head layout)
  #pragma unroll
  for (int v = 0; v < 8; ++v) {
    const int i = rowBase + v + 8 * half;
    const float inv = 1.0f / lloc[v];
    __bf16* out = Obuf + ((size_t)i * BATCH + m) * DMODEL + h * EHEAD;
    out[r]      = (__bf16)(o0[v] * inv);
    out[16 + r] = (__bf16)(o1[v] * inv);
    out[32 + r] = (__bf16)(o2[v] * inv);
    out[48 + r] = (__bf16)(o3[v] * inv);
  }
}

extern "C" void kernel_launch(void* const* d_in, const int* in_sizes, int n_in,
                              void* d_out, int out_size, void* d_ws, size_t ws_size,
                              hipStream_t stream) {
  (void)in_sizes; (void)n_in; (void)out_size; (void)ws_size;
  const float* query = (const float*)d_in[0];
  const float* key   = (const float*)d_in[1];
  const float* value = (const float*)d_in[2];
  const unsigned char* kmask = (const unsigned char*)d_in[3];
  const float* Wq = (const float*)d_in[4];
  const float* bq = (const float*)d_in[5];
  const float* Wk = (const float*)d_in[6];
  const float* bk = (const float*)d_in[7];
  const float* Wv = (const float*)d_in[8];
  const float* bv = (const float*)d_in[9];
  const float* Wo = (const float*)d_in[10];
  const float* bo = (const float*)d_in[11];

  // Workspace layout (~32MB + 16KB):
  const size_t nProj = (size_t)NHEAD * BATCH * S_LEN * EHEAD;   // 4M bf16 elems
  char* ws = (char*)d_ws;
  __bf16* Qh   = (__bf16*)ws;
  __bf16* Kh   = Qh + nProj;
  __bf16* Vh   = Kh + nProj;                                    // V^T layout
  __bf16* Obuf = Vh + nProj;                                    // [S*B x D] bf16
  float*  mb   = (float*)(Obuf + nProj);                        // [B x S] f32

  maskbias_kernel<<<dim3((S_LEN * BATCH + 255) / 256), dim3(256), 0, stream>>>(kmask, mb);

  const dim3 gg(DMODEL / 64, (S_LEN * BATCH) / 128);  // (16, 32)
  const dim3 bb(256);
  // q is pre-scaled by 1/sqrt(E)=0.125 so scores need no extra scaling
  gemm_bf16_wmma<false, 0><<<gg, bb, 0, stream>>>(query, Wq, bq, Qh, 0.125f);
  gemm_bf16_wmma<false, 0><<<gg, bb, 0, stream>>>(key,   Wk, bk, Kh, 1.0f);
  gemm_bf16_wmma<false, 2><<<gg, bb, 0, stream>>>(value, Wv, bv, Vh, 1.0f);

  const dim3 ga(S_LEN / 128, NHEAD * BATCH);          // (16, 32)
  attn_kernel<<<ga, bb, 0, stream>>>(Qh, Kh, Vh, mb, Obuf);

  gemm_bf16_wmma<true, 1><<<gg, bb, 0, stream>>>(Obuf, Wo, bo, d_out, 1.0f);
}
